// GAT_GCN_22539988369862
// MI455X (gfx1250) — matrix operand order repaired and checked by reading, not compiled, not run on hardware
//
#include <hip/hip_runtime.h>

// ---------------- problem constants (match reference) ----------------
#define N_NODES 50000
#define N_EDGES 800000
#define E_TOT   (N_EDGES + N_NODES)   // self-loops appended
#define F_IN    128
#define HID     32
#define HEADS   4
#define HID_ALL (HEADS * HID)         // 128
#define CLS     40
#define H2LD    48                    // padded leading dim for layer-2 features
#define NEG_SLOPE 0.2f

// ---------------- WMMA vector types ----------------
typedef __attribute__((ext_vector_type(16))) __bf16          bf16x16;
typedef __attribute__((ext_vector_type(16))) unsigned short  u16x16;
typedef __attribute__((ext_vector_type(8)))  float           floatx8;

// f32 -> bf16 with round-to-nearest-even
static __device__ __forceinline__ unsigned short f2bf(float f) {
    unsigned u = __builtin_bit_cast(unsigned, f);
    u += 0x7FFFu + ((u >> 16) & 1u);
    return (unsigned short)(u >> 16);
}
static __device__ __forceinline__ unsigned long long pack4bf(float a, float b,
                                                             float c, float d) {
    return (unsigned long long)f2bf(a)        | ((unsigned long long)f2bf(b) << 16) |
           ((unsigned long long)f2bf(c) << 32) | ((unsigned long long)f2bf(d) << 48);
}

// order-preserving float<->uint encoding (for atomicMax on floats)
static __device__ __forceinline__ unsigned encf(float f) {
    unsigned u = __builtin_bit_cast(unsigned, f);
    return (u & 0x80000000u) ? ~u : (u | 0x80000000u);
}
static __device__ __forceinline__ float decf(unsigned u) {
    unsigned v = (u & 0x80000000u) ? (u & 0x7FFFFFFFu) : ~u;
    return __builtin_bit_cast(float, v);
}

static __device__ __forceinline__ float lrelu(float a) {
    return a > 0.f ? a : NEG_SLOPE * a;
}

// A-matrix K permutation (ISA 16-bit A 16x32 layout):
//   lane half hf holds K in {hf*8 .. hf*8+7} U {16+hf*8 .. 23+hf*8}
//   given k: hf = (k>>3)&1 ; j = (k&7) + ((k>>4)<<3)
// B-matrix: same K permutation as A (same hf and j for each k).

// ---------------- init: zero accumulators, seed output with bias2 ----------------
__global__ void k_init(float* __restrict__ agg1, unsigned* __restrict__ amax1,
                       float* __restrict__ den1, unsigned* __restrict__ amax2,
                       float* __restrict__ den2, float* __restrict__ out,
                       const float* __restrict__ b2) {
    int i = blockIdx.x * blockDim.x + threadIdx.x;
    if (i >= N_NODES * HID_ALL) return;
    agg1[i] = 0.f;
    if (i < N_NODES * HEADS) { amax1[i] = 0u; den1[i] = 0.f; }
    if (i < N_NODES)         { amax2[i] = 0u; den2[i] = 0.f; }
    if (i < N_NODES * CLS)   { out[i] = b2[i % CLS]; }
}

// ---------------- GEMM1: h1[N,128] = x[N,128] @ W1[128,128], bf16 WMMA ----------------
// 256 threads = 8 waves; block tile 64 rows x 32 cols (wave grid 4x2), K-loop 4 x 32.
// LDS is fragment-major: each lane's 16-element fragment is one contiguous 32B chunk.
__global__ __launch_bounds__(256) void k_gemm1(const float* __restrict__ X,
                                               const float* __restrict__ W,
                                               float* __restrict__ H) {
    __shared__ alignas(32) unsigned short As[4 * 32 * 16]; // [rowTile][lane][j]
    __shared__ alignas(32) unsigned short Bs[2 * 32 * 16]; // [colTile][lane][j]
    const int t = threadIdx.x;
    const int lane = t & 31, wid = t >> 5;
    const int waveR = wid & 3, waveC = wid >> 2;
    const int rowBase = blockIdx.x * 64;
    const int colBase = blockIdx.y * 32;
    const int r16 = lane & 15, hf = lane >> 4;
    floatx8 acc = {0.f, 0.f, 0.f, 0.f, 0.f, 0.f, 0.f, 0.f};

#pragma unroll
    for (int kb = 0; kb < 4; ++kb) {
        const int k0 = kb * 32;
        // ---- stage A: 64x32 elements = 512 float4-chunks, 2 per thread ----
#pragma unroll
        for (int i = 0; i < 2; ++i) {
            int q = t + i * 256;           // chunk id
            int r = q >> 3;                // row within block tile, 0..63
            int k = (q & 7) << 2;          // k offset 0..28, 4-aligned
            int row = rowBase + r;
            int rowc = row < N_NODES ? row : N_NODES - 1;   // clamp: branchless guard
            const float4 v = *(const float4*)(X + (size_t)rowc * F_IN + k0 + k);
            float x0 = v.x, x1 = v.y, x2 = v.z, x3 = v.w;
            if (row >= N_NODES) { x0 = x1 = x2 = x3 = 0.f; }
            int hfk = (k >> 3) & 1;
            int j   = (k & 7) + ((k >> 4) << 3);
            int dl  = hfk * 16 + (r & 15);
            int tile = r >> 4;
            *(unsigned long long*)&As[(tile * 32 + dl) * 16 + j] = pack4bf(x0, x1, x2, x3);
        }
        // ---- stage B: 32x32 elements = 256 chunks (4 consecutive K, fixed col) ----
        {
            int c  = t & 31;               // column within block tile
            int kk = (t >> 5) << 2;        // K base 0..28, 4-aligned
            const float* wp = W + (size_t)(k0 + kk) * HID_ALL + colBase + c;
            int hfk = (kk >> 3) & 1;
            int j   = (kk & 7) + ((kk >> 4) << 3);
            int dl  = hfk * 16 + (c & 15);
            int tile = c >> 4;
            *(unsigned long long*)&Bs[(tile * 32 + dl) * 16 + j] =
                pack4bf(wp[0], wp[HID_ALL], wp[2 * HID_ALL], wp[3 * HID_ALL]);
        }
        __syncthreads();
        u16x16 au = *(const u16x16*)&As[(waveR * 32 + lane) * 16];
        u16x16 bu = *(const u16x16*)&Bs[(waveC * 32 + lane) * 16];
        bf16x16 a = __builtin_bit_cast(bf16x16, au);
        bf16x16 b = __builtin_bit_cast(bf16x16, bu);
        acc = __builtin_amdgcn_wmma_f32_16x16x32_bf16(false, a, false, b,
                                                      (short)0, acc, false, false);
        __syncthreads();
    }
    // Epilogue: N_NODES % 16 == 0, so each wave's 16-row output tile is either
    // fully valid or fully invalid -> one wave-uniform guard, unguarded stores.
    if (rowBase + waveR * 16 < N_NODES) {
        int rowT = rowBase + waveR * 16 + hf * 8;
        int col  = colBase + waveC * 16 + r16;
#pragma unroll
        for (int i = 0; i < 8; ++i) {      // C/D: VGPR i -> row hf*8+i, col = lane&15
            H[(rowT + i) * HID_ALL + col] = acc[i];
        }
    }
}

// ---------------- GEMM2: h2[N,48(pad)] = act1[N,128] @ W2[128,40], bf16 WMMA ----------------
// 192 threads = 6 waves; block tile 32 rows x 48 cols (pad 40->48), wave grid 2x3.
// Output stored with leading dim 48: no per-lane column guard needed; pad cols never read.
__global__ __launch_bounds__(192) void k_gemm2(const float* __restrict__ X,
                                               const float* __restrict__ W,
                                               float* __restrict__ H) {
    __shared__ alignas(32) unsigned short As[2 * 32 * 16]; // [rowTile][lane][j]
    __shared__ alignas(32) unsigned short Bs[3 * 32 * 16]; // [colTile][lane][j]
    const int t = threadIdx.x;
    const int lane = t & 31, wid = t >> 5;        // 0..5
    const int waveR = wid & 1, waveC = wid >> 1;  // 2 x 3
    const int rowBase = blockIdx.x * 32;
    const int r16 = lane & 15, hf = lane >> 4;
    floatx8 acc = {0.f, 0.f, 0.f, 0.f, 0.f, 0.f, 0.f, 0.f};

#pragma unroll
    for (int kb = 0; kb < 4; ++kb) {
        const int k0 = kb * 32;
        // ---- stage A: 32x32 = 256 float4-chunks ----
        for (int q = t; q < 256; q += 192) {
            int r = q >> 3;
            int k = (q & 7) << 2;
            int row = rowBase + r;
            int rowc = row < N_NODES ? row : N_NODES - 1;
            const float4 v = *(const float4*)(X + (size_t)rowc * F_IN + k0 + k);
            float x0 = v.x, x1 = v.y, x2 = v.z, x3 = v.w;
            if (row >= N_NODES) { x0 = x1 = x2 = x3 = 0.f; }
            int hfk = (k >> 3) & 1;
            int j   = (k & 7) + ((k >> 4) << 3);
            int dl  = hfk * 16 + (r & 15);
            int tile = r >> 4;
            *(unsigned long long*)&As[(tile * 32 + dl) * 16 + j] = pack4bf(x0, x1, x2, x3);
        }
        // ---- stage B: 32x48 = 384 chunks (4 consecutive K, fixed col) ----
        for (int q = t; q < 384; q += 192) {
            int c  = q % 48;
            int kk = (q / 48) << 2;
            int cc = c < CLS ? c : CLS - 1;                  // clamp: branchless guard
            const float* wp = W + (size_t)(k0 + kk) * CLS + cc;
            float x0 = wp[0], x1 = wp[CLS], x2 = wp[2 * CLS], x3 = wp[3 * CLS];
            if (c >= CLS) { x0 = x1 = x2 = x3 = 0.f; }
            int hfk = (kk >> 3) & 1;
            int j   = (kk & 7) + ((kk >> 4) << 3);
            int dl  = hfk * 16 + (c & 15);
            int tile = c >> 4;
            *(unsigned long long*)&Bs[(tile * 32 + dl) * 16 + j] = pack4bf(x0, x1, x2, x3);
        }
        __syncthreads();
        u16x16 au = *(const u16x16*)&As[(waveR * 32 + lane) * 16];
        u16x16 bu = *(const u16x16*)&Bs[(waveC * 32 + lane) * 16];
        bf16x16 a = __builtin_bit_cast(bf16x16, au);
        bf16x16 b = __builtin_bit_cast(bf16x16, bu);
        acc = __builtin_amdgcn_wmma_f32_16x16x32_bf16(false, a, false, b,
                                                      (short)0, acc, false, false);
        __syncthreads();
    }
    if (rowBase + waveR * 16 < N_NODES) {      // N_NODES % 16 == 0: tile all-or-nothing
        int rowT = rowBase + waveR * 16 + hf * 8;
        int col  = waveC * 16 + r16;           // 0..47, stored with ld=48, no guard
#pragma unroll
        for (int i = 0; i < 8; ++i) {
            H[(rowT + i) * H2LD + col] = acc[i];
        }
    }
}

// ---------------- attention scores, layer 1: thread per (node, head) ----------------
__global__ void k_attn1(const float* __restrict__ H, const float* __restrict__ att_s,
                        const float* __restrict__ att_d,
                        float* __restrict__ as_, float* __restrict__ ad_) {
    int i = blockIdx.x * blockDim.x + threadIdx.x;   // n*HEADS + h
    if (i >= N_NODES * HEADS) return;
    int n = i >> 2, h = i & 3;
    const float* hp = H + n * HID_ALL + h * HID;
    float s = 0.f, d = 0.f;
#pragma unroll
    for (int c = 0; c < HID; ++c) {
        float v = hp[c];
        s += v * att_s[h * HID + c];
        d += v * att_d[h * HID + c];
    }
    as_[i] = s; ad_[i] = d;
}

// ---------------- attention scores, layer 2: thread per node ----------------
__global__ void k_attn2(const float* __restrict__ H2, const float* __restrict__ att_s,
                        const float* __restrict__ att_d,
                        float* __restrict__ as_, float* __restrict__ ad_) {
    int n = blockIdx.x * blockDim.x + threadIdx.x;
    if (n >= N_NODES) return;
    float s = 0.f, d = 0.f;
#pragma unroll
    for (int c = 0; c < CLS; ++c) {
        float v = H2[n * H2LD + c];
        s += v * att_s[c];
        d += v * att_d[c];
    }
    as_[n] = s; ad_[n] = d;
}

// ---------------- edge segment-max (encoded float atomicMax in L2) ----------------
__global__ void k_edge_max(const int* __restrict__ esrc, const int* __restrict__ edst,
                           const float* __restrict__ as_, const float* __restrict__ ad_,
                           unsigned* __restrict__ amax, int heads) {
    int e = blockIdx.x * blockDim.x + threadIdx.x;
    if (e >= E_TOT) return;
    int s = (e < N_EDGES) ? esrc[e] : (e - N_EDGES);
    int d = (e < N_EDGES) ? edst[e] : (e - N_EDGES);
    for (int h = 0; h < heads; ++h) {
        float a = lrelu(as_[s * heads + h] + ad_[d * heads + h]);
        atomicMax(&amax[d * heads + h], encf(a));
    }
}

// ---------------- edge softmax denominator (atomic f32 add in L2) ----------------
__global__ void k_edge_sum(const int* __restrict__ esrc, const int* __restrict__ edst,
                           const float* __restrict__ as_, const float* __restrict__ ad_,
                           const unsigned* __restrict__ amax, float* __restrict__ den,
                           int heads) {
    int e = blockIdx.x * blockDim.x + threadIdx.x;
    if (e >= E_TOT) return;
    int s = (e < N_EDGES) ? esrc[e] : (e - N_EDGES);
    int d = (e < N_EDGES) ? edst[e] : (e - N_EDGES);
    for (int h = 0; h < heads; ++h) {
        float a = lrelu(as_[s * heads + h] + ad_[d * heads + h]);
        float ex = expf(a - decf(amax[d * heads + h]));
        atomicAdd(&den[d * heads + h], ex);
    }
}

// ---------------- layer-1 weighted scatter: one wave32 per edge, lanes = channels ----------------
__global__ __launch_bounds__(256) void k_scatter1(const int* __restrict__ esrc,
                                                  const int* __restrict__ edst,
                                                  const float* __restrict__ as_,
                                                  const float* __restrict__ ad_,
                                                  const unsigned* __restrict__ amax,
                                                  const float* __restrict__ den,
                                                  const float* __restrict__ H,
                                                  float* __restrict__ agg) {
    int lane = threadIdx.x & 31;
    int e = blockIdx.x * 8 + (threadIdx.x >> 5);
    if (e >= E_TOT) return;
    int s = (e < N_EDGES) ? esrc[e] : (e - N_EDGES);
    int d = (e < N_EDGES) ? edst[e] : (e - N_EDGES);
#pragma unroll
    for (int h = 0; h < HEADS; ++h) {
        float a = lrelu(as_[s * HEADS + h] + ad_[d * HEADS + h]);
        float coef = expf(a - decf(amax[d * HEADS + h])) / (den[d * HEADS + h] + 1e-16f);
        float v = H[s * HID_ALL + h * HID + lane] * coef;
        atomicAdd(&agg[d * HID_ALL + h * HID + lane], v);
    }
}

// ---------------- bias + ELU (in place: agg1 becomes act1) ----------------
__global__ void k_finish1(float* __restrict__ agg, const float* __restrict__ b1) {
    int i = blockIdx.x * blockDim.x + threadIdx.x;
    if (i >= N_NODES * HID_ALL) return;
    float v = agg[i] + b1[i & (HID_ALL - 1)];
    agg[i] = v > 0.f ? v : (expf(v) - 1.f);
}

// ---------------- layer-2 weighted scatter into d_out (pre-seeded with bias2) ----------------
__global__ __launch_bounds__(256) void k_scatter2(const int* __restrict__ esrc,
                                                  const int* __restrict__ edst,
                                                  const float* __restrict__ as_,
                                                  const float* __restrict__ ad_,
                                                  const unsigned* __restrict__ amax,
                                                  const float* __restrict__ den,
                                                  const float* __restrict__ H2,
                                                  float* __restrict__ out) {
    int lane = threadIdx.x & 31;
    int e = blockIdx.x * 8 + (threadIdx.x >> 5);
    if (e >= E_TOT) return;
    int s = (e < N_EDGES) ? esrc[e] : (e - N_EDGES);
    int d = (e < N_EDGES) ? edst[e] : (e - N_EDGES);
    float a = lrelu(as_[s] + ad_[d]);
    float coef = expf(a - decf(amax[d])) / (den[d] + 1e-16f);
    for (int c = lane; c < CLS; c += 32) {
        atomicAdd(&out[d * CLS + c], H2[s * H2LD + c] * coef);
    }
}

// ---------------- host launch ----------------
extern "C" void kernel_launch(void* const* d_in, const int* in_sizes, int n_in,
                              void* d_out, int out_size, void* d_ws, size_t ws_size,
                              hipStream_t stream) {
    const float* x        = (const float*)d_in[0];
    const int*   ei       = (const int*)d_in[1];     // [2, E] flat
    const float* W1       = (const float*)d_in[2];
    const float* att_src1 = (const float*)d_in[3];
    const float* att_dst1 = (const float*)d_in[4];
    const float* b1       = (const float*)d_in[5];
    const float* W2       = (const float*)d_in[6];
    const float* att_src2 = (const float*)d_in[7];
    const float* att_dst2 = (const float*)d_in[8];
    const float* b2       = (const float*)d_in[9];
    float* out = (float*)d_out;

    const int* esrc = ei;
    const int* edst = ei + N_EDGES;

    // workspace carve (~65 MB total; all L2-resident on MI455X's 192 MB L2)
    float*    h1    = (float*)d_ws;                       // N*128
    float*    agg1  = h1 + (size_t)N_NODES * HID_ALL;     // N*128 (becomes act1)
    float*    asrc1 = agg1 + (size_t)N_NODES * HID_ALL;   // N*4
    float*    adst1 = asrc1 + (size_t)N_NODES * HEADS;    // N*4
    unsigned* amax1 = (unsigned*)(adst1 + (size_t)N_NODES * HEADS);   // N*4
    float*    den1  = (float*)(amax1 + (size_t)N_NODES * HEADS);      // N*4
    float*    h2    = den1 + (size_t)N_NODES * HEADS;     // N*48 (padded ld)
    float*    asrc2 = h2 + (size_t)N_NODES * H2LD;        // N
    float*    adst2 = asrc2 + N_NODES;                    // N
    unsigned* amax2 = (unsigned*)(adst2 + N_NODES);       // N
    float*    den2  = (float*)(amax2 + N_NODES);          // N

    const int TPB = 256;
    const int gNF   = (N_NODES * HID_ALL + TPB - 1) / TPB;
    const int gNH   = (N_NODES * HEADS + TPB - 1) / TPB;
    const int gN    = (N_NODES + TPB - 1) / TPB;
    const int gE    = (E_TOT + TPB - 1) / TPB;
    const int gEw   = (E_TOT + 7) / 8;       // wave-per-edge kernels

    k_init<<<gNF, TPB, 0, stream>>>(agg1, amax1, den1, amax2, den2, out, b2);

    // ---- layer 1 ----
    k_gemm1<<<dim3((N_NODES + 63) / 64, HID_ALL / 32), 256, 0, stream>>>(x, W1, h1);
    k_attn1<<<gNH, TPB, 0, stream>>>(h1, att_src1, att_dst1, asrc1, adst1);
    k_edge_max<<<gE, TPB, 0, stream>>>(esrc, edst, asrc1, adst1, amax1, HEADS);
    k_edge_sum<<<gE, TPB, 0, stream>>>(esrc, edst, asrc1, adst1, amax1, den1, HEADS);
    k_scatter1<<<gEw, 256, 0, stream>>>(esrc, edst, asrc1, adst1, amax1, den1, h1, agg1);
    k_finish1<<<gNF, TPB, 0, stream>>>(agg1, b1);

    // ---- layer 2 ----
    k_gemm2<<<dim3((N_NODES + 31) / 32), 192, 0, stream>>>(agg1, W2, h2);
    k_attn2<<<gN, TPB, 0, stream>>>(h2, att_src2, att_dst2, asrc2, adst2);
    k_edge_max<<<gE, TPB, 0, stream>>>(esrc, edst, asrc2, adst2, amax2, 1);
    k_edge_sum<<<gE, TPB, 0, stream>>>(esrc, edst, asrc2, adst2, amax2, den2, 1);
    k_scatter2<<<gEw, 256, 0, stream>>>(esrc, edst, asrc2, adst2, amax2, den2, h2, out);
}
